// Net_85890755985736
// MI455X (gfx1250) — compile-verified, matrix operand order checked by reading, not run on hardware
//
#include <hip/hip_runtime.h>

#define N_VOX 400000
#define KOFF  27

typedef __attribute__((ext_vector_type(16))) _Float16 v16h;
typedef __attribute__((ext_vector_type(8)))  float    v8f;

// Channel permutation that makes per-lane A-fragments contiguous:
// swap bits 3 and 4 of the channel index. Applied when WRITING activations;
// layer inputs are then loaded as a single 32B v16h per lane per 32-K step.
__device__ __host__ inline int perm_ch(int c) {
  return (c & ~0x18) | ((c & 8) << 1) | ((c & 16) >> 1);
}

// ---------------------------------------------------------------------------
// Weight pre-swizzle: W [27, cin, cout] f32 row-major ->
// B-fragments f16: [27][cout/16][ksteps][32 lanes][16 halfs]; each lane's 16
// halfs contiguous (one 32B coalesced load per lane at WMMA time).
// B (KxN=32x16): lanes 0-15 hold col=lane, K rows s*32+0..15; lanes 16-31
// hold col=lane-16, K rows s*32+16..31. For cin=16 rows 8-15, 24-31 are zero.
// For cin=64 the fragment-K axis indexes the *permuted* channel order; for
// cin=16 it matches the featsh embedded-zero layout (c at c<8 ? c : c+8).
// ---------------------------------------------------------------------------
__global__ void swizzle_w_kernel(const float* __restrict__ W,
                                 _Float16* __restrict__ Wf,
                                 int cin, int cout, int ksteps) {
  int t = blockIdx.x * blockDim.x + threadIdx.x;
  int total = KOFF * (cout / 16) * ksteps * 32;
  if (t >= total) return;
  int lane = t & 31;
  int rest = t >> 5;
  int s = rest % ksteps;
  rest /= ksteps;
  int n = rest % (cout / 16);
  int k = rest / (cout / 16);
  int col = n * 16 + (lane & 15);
  int rowbase = s * 32 + ((lane & 16) ? 16 : 0);
  _Float16* dst = Wf + (size_t)t * 16;
  for (int j = 0; j < 16; ++j) {
    int r = rowbase + j;  // fragment-K index
    float v = 0.0f;
    if (cin == 64) {
      v = W[((size_t)k * 64 + perm_ch(r)) * cout + col];  // perm is involution
    } else {
      int within = r & 31;
      int c = -1;
      if (within < 8) c = within;
      else if (within >= 16 && within < 24) c = within - 8;
      if (c >= 0 && r < 32) v = W[((size_t)k * 16 + c) * cout + col];
    }
    dst[j] = (_Float16)v;
  }
}

// feats f32 [N,16] -> f16 [N+1, 32] fragment-ready rows with embedded zero
// K-padding: position p holds channel (p<8 ? p : p-8) for p in [0,8)u[16,24),
// zero elsewhere. Row N_VOX is all zeros.
__global__ void feats_to_h_kernel(const float* __restrict__ f,
                                  _Float16* __restrict__ fh) {
  int t = blockIdx.x * blockDim.x + threadIdx.x;
  if (t >= (N_VOX + 1) * 32) return;
  int row = t >> 5;
  int pos = t & 31;
  float v = 0.0f;
  if (row < N_VOX) {
    int c = -1;
    if (pos < 8) c = pos;
    else if (pos >= 16 && pos < 24) c = pos - 8;
    if (c >= 0) v = f[row * 16 + c];
  }
  fh[t] = (_Float16)v;
}

// zero the padding row (index N_VOX) of the two 64-channel activation buffers
__global__ void zero_pad_kernel(_Float16* __restrict__ hA,
                                _Float16* __restrict__ hB) {
  int t = threadIdx.x;
  if (t < 64) {
    hA[(size_t)N_VOX * 64 + t] = (_Float16)0.0f;
    hB[(size_t)N_VOX * 64 + t] = (_Float16)0.0f;
  }
}

// W5 f32 [27,64,1] -> f16 [27*64], permuted to match the permuted activations
__global__ void w5_to_h_kernel(const float* __restrict__ w,
                               _Float16* __restrict__ wh) {
  int t = blockIdx.x * blockDim.x + threadIdx.x;
  if (t >= KOFF * 64) return;
  int k = t >> 6;
  int c = t & 63;
  wh[k * 64 + perm_ch(c)] = (_Float16)w[k * 64 + c];
}

// ---------------------------------------------------------------------------
// WMMA sparse-conv layer. One wave32 owns a 32-voxel M-tile (two 16-row
// subtiles sharing B-fragments) x 64 output channels = 8 f32 C-fragments.
// Steady state per offset: 2 nbr loads + 4 A loads + 8 B loads vs 16 WMMAs.
// N_VOX = 3125 blocks * 4 waves * 32 voxels exactly -> every tile full,
// EXEC all ones at every WMMA (ISA requirement).
// ---------------------------------------------------------------------------
template <int CIN>
__global__ __launch_bounds__(128) void sconv_wmma_kernel(
    const _Float16* __restrict__ fin,   // [N+1, CSTRIDE] fragment-ready
    const _Float16* __restrict__ Wf,    // [27][4][KSTEPS][32][16]
    const int* __restrict__ nbr,        // [N, 27]
    _Float16* __restrict__ fout) {      // [N+1, 64] (permuted channel order)
  constexpr int KSTEPS = (CIN == 16) ? 1 : 2;
  constexpr int CSTRIDE = (CIN == 16) ? 32 : 64;
  const int lane = threadIdx.x & 31;
  const int wave = threadIdx.x >> 5;
  const int base = (blockIdx.x * 4 + wave) * 32;   // tile's first voxel
  const int r0 = base + (lane & 15);               // subtile-0 gather row
  const int r1 = r0 + 16;                          // subtile-1 gather row
  const int aofs = (lane & 16) ? 16 : 0;           // fragment half-group

  const int* __restrict__ nb0 = nbr + (size_t)r0 * KOFF;
  const int* __restrict__ nb1 = nbr + (size_t)r1 * KOFF;

  v8f acc0[4], acc1[4];
#pragma unroll
  for (int n = 0; n < 4; ++n) { acc0[n] = (v8f){}; acc1[n] = (v8f){}; }

  for (int k = 0; k < KOFF; ++k) {
    int i0 = nb0[k]; i0 = (i0 < 0) ? N_VOX : i0;   // -1 -> zero row
    int i1 = nb1[k]; i1 = (i1 < 0) ? N_VOX : i1;
    const _Float16* a0p = fin + (size_t)i0 * CSTRIDE + aofs;
    const _Float16* a1p = fin + (size_t)i1 * CSTRIDE + aofs;

    // A-fragments: one 32B v16h load per lane per subtile per 32-K step
    v16h a00 = *(const v16h*)a0p;
    v16h a10 = *(const v16h*)a1p;
    v16h a01, a11;
    if constexpr (CIN == 64) {
      a01 = *(const v16h*)(a0p + 32);
      a11 = *(const v16h*)(a1p + 32);
    }

    const _Float16* wk = Wf + (size_t)k * 4 * KSTEPS * 32 * 16;
#pragma unroll
    for (int n = 0; n < 4; ++n) {
      const _Float16* wb = wk + ((size_t)(n * KSTEPS) * 32 + lane) * 16;
      v16h b0 = *(const v16h*)wb;                  // shared by both subtiles
      acc0[n] = __builtin_amdgcn_wmma_f32_16x16x32_f16(
          false, a00, false, b0, (short)0, acc0[n], false, false);
      acc1[n] = __builtin_amdgcn_wmma_f32_16x16x32_f16(
          false, a10, false, b0, (short)0, acc1[n], false, false);
      if constexpr (CIN == 64) {
        v16h b1 = *(const v16h*)(wb + 32 * 16);
        acc0[n] = __builtin_amdgcn_wmma_f32_16x16x32_f16(
            false, a01, false, b1, (short)0, acc0[n], false, false);
        acc1[n] = __builtin_amdgcn_wmma_f32_16x16x32_f16(
            false, a11, false, b1, (short)0, acc1[n], false, false);
      }
    }
  }

  // Fused ReLU + f16 convert + permuted-column store. D layout: VGPR v holds
  // M=v (lanes 0-15) / M=v+8 (lanes 16-31), N = lane&15.
  const int rofs = (lane & 16) ? 8 : 0;
  const int col = lane & 15;
#pragma unroll
  for (int n = 0; n < 4; ++n) {
    const int pcol = perm_ch(n * 16 + col);        // write permuted layout
#pragma unroll
    for (int v = 0; v < 8; ++v) {
      const int row = base + rofs + v;
      float x0 = acc0[n][v];
      x0 = x0 > 0.0f ? x0 : 0.0f;
      fout[(size_t)row * 64 + pcol] = (_Float16)x0;
      float x1 = acc1[n][v];
      x1 = x1 > 0.0f ? x1 : 0.0f;
      fout[(size_t)(row + 16) * 64 + pcol] = (_Float16)x1;
    }
  }
}

// ---------------------------------------------------------------------------
// Final 64 -> 1 layer (0.5% of total FLOPs): per-voxel VALU dot. Both the
// activations and w5h are in the same permuted channel order, so the dot
// product is order-invariant.
// ---------------------------------------------------------------------------
__global__ __launch_bounds__(256) void sconv_out_kernel(
    const _Float16* __restrict__ fin,   // [N+1, 64] f16 (permuted)
    const _Float16* __restrict__ w5,    // [27*64] f16 (permuted)
    const int* __restrict__ nbr,
    float* __restrict__ out) {
  int t = blockIdx.x * blockDim.x + threadIdx.x;
  if (t >= N_VOX) return;
  float acc = 0.0f;
  for (int k = 0; k < KOFF; ++k) {
    int idx = nbr[(size_t)t * KOFF + k];
    if (idx < 0) continue;
    const _Float16* h = fin + (size_t)idx * 64;
    const _Float16* w = w5 + k * 64;
#pragma unroll 8
    for (int c = 0; c < 64; ++c) acc += (float)h[c] * (float)w[c];
  }
  out[t] = acc;
}

// ---------------------------------------------------------------------------
extern "C" void kernel_launch(void* const* d_in, const int* in_sizes, int n_in,
                              void* d_out, int out_size, void* d_ws, size_t ws_size,
                              hipStream_t stream) {
  const float* feats = (const float*)d_in[0];
  const float* W1 = (const float*)d_in[1];
  const float* W2 = (const float*)d_in[2];
  const float* W3 = (const float*)d_in[3];
  const float* W4 = (const float*)d_in[4];
  const float* W5 = (const float*)d_in[5];
  const int* nbr = (const int*)d_in[6];
  float* out = (float*)d_out;

  // Workspace carve-up (256B aligned)
  char* ws = (char*)d_ws;
  size_t off = 0;
  auto alloc = [&](size_t bytes) {
    char* p = ws + off;
    off += (bytes + 255) & ~(size_t)255;
    return p;
  };
  _Float16* featsh = (_Float16*)alloc((size_t)(N_VOX + 1) * 32 * 2);
  _Float16* hA     = (_Float16*)alloc((size_t)(N_VOX + 1) * 64 * 2);
  _Float16* hB     = (_Float16*)alloc((size_t)(N_VOX + 1) * 64 * 2);
  _Float16* Wf1    = (_Float16*)alloc((size_t)KOFF * 4 * 1 * 32 * 16 * 2);
  _Float16* Wf2    = (_Float16*)alloc((size_t)KOFF * 4 * 2 * 32 * 16 * 2);
  _Float16* Wf3    = (_Float16*)alloc((size_t)KOFF * 4 * 2 * 32 * 16 * 2);
  _Float16* Wf4    = (_Float16*)alloc((size_t)KOFF * 4 * 2 * 32 * 16 * 2);
  _Float16* w5h    = (_Float16*)alloc((size_t)KOFF * 64 * 2);
  (void)ws_size; (void)in_sizes; (void)n_in; (void)out_size;

  // --- preprocessing ---
  {
    int total = (N_VOX + 1) * 32;
    feats_to_h_kernel<<<(total + 255) / 256, 256, 0, stream>>>(feats, featsh);
  }
  zero_pad_kernel<<<1, 64, 0, stream>>>(hA, hB);
  {
    int t1 = KOFF * 4 * 1 * 32;
    swizzle_w_kernel<<<(t1 + 255) / 256, 256, 0, stream>>>(W1, Wf1, 16, 64, 1);
    int t2 = KOFF * 4 * 2 * 32;
    swizzle_w_kernel<<<(t2 + 255) / 256, 256, 0, stream>>>(W2, Wf2, 64, 64, 2);
    swizzle_w_kernel<<<(t2 + 255) / 256, 256, 0, stream>>>(W3, Wf3, 64, 64, 2);
    swizzle_w_kernel<<<(t2 + 255) / 256, 256, 0, stream>>>(W4, Wf4, 64, 64, 2);
    w5_to_h_kernel<<<(KOFF * 64 + 255) / 256, 256, 0, stream>>>(W5, w5h);
  }

  // --- 4 WMMA conv layers: 12500 M-tiles of 32 voxels, 4 waves per block
  //     -> 128 voxels per block, grid = 400000 / 128 = 3125 (exact)
  static_assert(N_VOX % 128 == 0, "tiles must be exact");
  const int grid = N_VOX / 128;
  sconv_wmma_kernel<16><<<grid, 128, 0, stream>>>(featsh, Wf1, nbr, hA);
  sconv_wmma_kernel<64><<<grid, 128, 0, stream>>>(hA, Wf2, nbr, hB);
  sconv_wmma_kernel<64><<<grid, 128, 0, stream>>>(hB, Wf3, nbr, hA);
  sconv_wmma_kernel<64><<<grid, 128, 0, stream>>>(hA, Wf4, nbr, hB);

  // --- final 64 -> 1 layer ---
  sconv_out_kernel<<<(N_VOX + 255) / 256, 256, 0, stream>>>(hB, w5h, nbr, out);
}
